// GraphIsomorphismEdge_62483184222837
// MI455X (gfx1250) — matrix operand order repaired and checked by reading, not compiled, not run on hardware
//
#include <hip/hip_runtime.h>
#include <hip/hip_bf16.h>

typedef __attribute__((ext_vector_type(16))) __bf16 v16bf;
typedef __attribute__((ext_vector_type(8)))  __bf16 v8bf;
typedef __attribute__((ext_vector_type(8)))  float  v8f;

#define D_FEAT 128
#define H_FEAT 512

// ---------------------------------------------------------------- zero u ----
__global__ void zero_u(float4* u4, long n4) {
    long i = (long)blockIdx.x * blockDim.x + threadIdx.x;
    if (i < n4) u4[i] = make_float4(0.f, 0.f, 0.f, 0.f);
}

// ------------------------------------------- convert + transpose W1 / W2 ----
// W1 [128][512] f32  -> w1t [512][128] bf16   (row = output col n, contiguous K)
// W2 [512][128] f32  -> w2t [128][512] bf16
__global__ void prep_w(const float* __restrict__ W1, const float* __restrict__ W2,
                       __bf16* __restrict__ w1t, __bf16* __restrict__ w2t) {
    int i = blockIdx.x * blockDim.x + threadIdx.x;
    if (i < D_FEAT * H_FEAT) {
        int k1 = i >> 9, n1 = i & 511;               // W1: i = k*512 + n
        w1t[(size_t)n1 * D_FEAT + k1] = (__bf16)W1[i];
        int k2 = i >> 7, n2 = i & 127;               // W2: i = k*128 + n
        w2t[(size_t)n2 * H_FEAT + k2] = (__bf16)W2[i];
    }
}

// ----------------------------------------------------- edge scatter-add -----
// One wave per edge; lane l owns floats [4l, 4l+4) of the 128-wide row.
__global__ void scatter_edges(const float* __restrict__ h, const float* __restrict__ e_ij,
                              const int* __restrict__ src, const int* __restrict__ dst,
                              float* __restrict__ u, int E) {
    int wave = (int)((blockIdx.x * (unsigned)blockDim.x + threadIdx.x) >> 5);
    int lane = threadIdx.x & 31;
    if (wave >= E) return;
    int s = src[wave];                  // wave-uniform -> scalar loads
    int d = dst[wave];
    float4 hv = *(const float4*)(h    + (size_t)s    * D_FEAT + lane * 4);
    float4 ev = *(const float4*)(e_ij + (size_t)wave * D_FEAT + lane * 4);
    float* up = u + (size_t)d * D_FEAT + lane * 4;
    atomicAdd(up + 0, hv.x + ev.x);
    atomicAdd(up + 1, hv.y + ev.y);
    atomicAdd(up + 2, hv.z + ev.z);
    atomicAdd(up + 3, hv.w + ev.w);
}

// ---------------------- fused MLP (WMMA bf16) + residual + LayerNorm --------
// Block = 256 threads (8 waves) handles 16 nodes.
__launch_bounds__(256, 2)
__global__ void fused_mlp_ln(const float* __restrict__ u, const float* __restrict__ h,
                             const __bf16* __restrict__ w1t, const __bf16* __restrict__ w2t,
                             const float* __restrict__ b1, const float* __restrict__ b2,
                             const float* __restrict__ gamma, const float* __restrict__ beta,
                             float* __restrict__ out, int N) {
    __shared__ __bf16 hidS[16][H_FEAT];   // 16 KB: relu(u@W1+b1) as bf16
    __shared__ float  oS[16][D_FEAT];     //  8 KB: hid@W2 (pre residual/LN)

    const int node0 = blockIdx.x * 16;
    const int lane  = threadIdx.x & 31;
    const int w     = threadIdx.x >> 5;        // wave id 0..7
    const int mlan  = lane & 15;               // M (A/C) or N (B) lane index
    const int half  = lane >> 4;               // K-half selector
    const int mbase = half * 8;                // C-tile M base for this half

    // ---- A fragments for GEMM1: rows = nodes, K = 0..127 (4 steps of 32) ----
    int arow_i = node0 + mlan; if (arow_i > N - 1) arow_i = N - 1;
    const float* arow = u + (size_t)arow_i * D_FEAT;
    v16bf afrag[4];
#pragma unroll
    for (int kk = 0; kk < 4; ++kk) {
        int k0 = kk * 32 + half * 16;          // elems 0..15 at k0..k0+15
        float4 f0 = *(const float4*)(arow + k0);
        float4 f1 = *(const float4*)(arow + k0 + 4);
        float4 f2 = *(const float4*)(arow + k0 + 8);
        float4 f3 = *(const float4*)(arow + k0 + 12);
        v16bf a;
        a[0]=(__bf16)f0.x; a[1]=(__bf16)f0.y; a[2]=(__bf16)f0.z; a[3]=(__bf16)f0.w;
        a[4]=(__bf16)f1.x; a[5]=(__bf16)f1.y; a[6]=(__bf16)f1.z; a[7]=(__bf16)f1.w;
        a[8]=(__bf16)f2.x; a[9]=(__bf16)f2.y; a[10]=(__bf16)f2.z; a[11]=(__bf16)f2.w;
        a[12]=(__bf16)f3.x; a[13]=(__bf16)f3.y; a[14]=(__bf16)f3.z; a[15]=(__bf16)f3.w;
        afrag[kk] = a;
    }

    // ---- GEMM1: wave w computes hidden cols [w*64, w*64+64) ----
#pragma unroll
    for (int nt = 0; nt < 4; ++nt) {
        int nn = w * 64 + nt * 16 + mlan;      // B lane column
        v8f acc = {};
#pragma unroll
        for (int kk = 0; kk < 4; ++kk) {
            int kb = kk * 32 + half * 16;      // 16 consecutive K for this lane
            v16bf b = *(const v16bf*)(w1t + (size_t)nn * D_FEAT + kb);
            acc = __builtin_amdgcn_wmma_f32_16x16x32_bf16(
                      false, afrag[kk], false, b, (short)0, acc, false, false);
        }
        float b1v = b1[nn];
#pragma unroll
        for (int r = 0; r < 8; ++r) {          // C: VGPR r -> M = mbase+r, N = mlan
            float v = acc[r] + b1v;
            hidS[mbase + r][nn] = (__bf16)(v > 0.f ? v : 0.f);
        }
    }
    __syncthreads();

    // ---- GEMM2: wave w computes out cols [w*16, w*16+16), K = 512 ----
    {
        int nn = w * 16 + mlan;
        v8f acc = {};
#pragma unroll
        for (int kk = 0; kk < 16; ++kk) {
            int k0 = kk * 32 + half * 16;
            v8bf lo = *(const v8bf*)(&hidS[mlan][k0]);
            v8bf hi = *(const v8bf*)(&hidS[mlan][k0 + 8]);
            v16bf a;
#pragma unroll
            for (int e = 0; e < 8; ++e) { a[e] = lo[e]; a[8 + e] = hi[e]; }
            int kb = kk * 32 + half * 16;
            v16bf b = *(const v16bf*)(w2t + (size_t)nn * H_FEAT + kb);
            acc = __builtin_amdgcn_wmma_f32_16x16x32_bf16(
                      false, a, false, b, (short)0, acc, false, false);
        }
#pragma unroll
        for (int r = 0; r < 8; ++r) oS[mbase + r][nn] = acc[r];
    }
    __syncthreads();

    // ---- residual + bias + LayerNorm (16 threads per row, 8 cols each) ----
    {
        int row = threadIdx.x >> 4;            // 0..15
        int cg  = threadIdx.x & 15;
        int c0  = cg * 8;
        int node = node0 + row;
        bool valid = node < N;
        const float* hrow = h + (size_t)(valid ? node : N - 1) * D_FEAT;
        float v[8]; float s = 0.f, ss = 0.f;
#pragma unroll
        for (int j = 0; j < 8; ++j) {
            int c = c0 + j;
            float x = oS[row][c] + b2[c] + hrow[c];
            v[j] = x; s += x; ss += x * x;
        }
#pragma unroll
        for (int m = 1; m < 16; m <<= 1) {     // reduce within 16-lane half
            s  += __shfl_xor(s,  m, 32);
            ss += __shfl_xor(ss, m, 32);
        }
        float mean = s * (1.f / 128.f);
        float var  = ss * (1.f / 128.f) - mean * mean;
        float inv  = rsqrtf(var + 1e-5f);
        if (valid) {
            float* orow = out + (size_t)node * D_FEAT;
#pragma unroll
            for (int j = 0; j < 8; ++j) {
                int c = c0 + j;
                orow[c] = (v[j] - mean) * inv * gamma[c] + beta[c];
            }
        }
    }
}

// ---------------------------------------------------------------------------
extern "C" void kernel_launch(void* const* d_in, const int* in_sizes, int n_in,
                              void* d_out, int out_size, void* d_ws, size_t ws_size,
                              hipStream_t stream) {
    const float* h     = (const float*)d_in[0];
    const float* e_ij  = (const float*)d_in[1];
    const int*   src   = (const int*)d_in[2];
    const int*   dst   = (const int*)d_in[3];
    const float* W1    = (const float*)d_in[4];
    const float* b1    = (const float*)d_in[5];
    const float* W2    = (const float*)d_in[6];
    const float* b2    = (const float*)d_in[7];
    const float* gamma = (const float*)d_in[8];
    const float* beta  = (const float*)d_in[9];
    float* out = (float*)d_out;

    const int N = in_sizes[0] / D_FEAT;
    const int E = in_sizes[2];

    // workspace layout: u [N*128 f32] | w1t [512*128 bf16] | w2t [128*512 bf16]
    float* u = (float*)d_ws;
    size_t uBytes = ((size_t)N * D_FEAT * sizeof(float) + 255) & ~(size_t)255;
    __bf16* w1t = (__bf16*)((char*)d_ws + uBytes);
    __bf16* w2t = w1t + (size_t)D_FEAT * H_FEAT;

    long n4 = (long)N * D_FEAT / 4;
    zero_u<<<dim3((unsigned)((n4 + 255) / 256)), 256, 0, stream>>>((float4*)u, n4);

    prep_w<<<dim3((D_FEAT * H_FEAT + 255) / 256), 256, 0, stream>>>(W1, W2, w1t, w2t);

    long sthreads = (long)E * 32;
    scatter_edges<<<dim3((unsigned)((sthreads + 255) / 256)), 256, 0, stream>>>(
        h, e_ij, src, dst, u, E);

    fused_mlp_ln<<<dim3((N + 15) / 16), 256, 0, stream>>>(
        u, h, w1t, w2t, b1, b2, gamma, beta, out, N);
}
